// GRASPEncoder_5188320493816
// MI455X (gfx1250) — compile-verified, matrix operand order checked by reading, not run on hardware
//
#include <hip/hip_runtime.h>
#include <math.h>

// ---------------------------------------------------------------------------
// GRASP encoder for MI455X (gfx1250, wave32, WMMA).
// Heavy GEMMs (x@w_ih^T : 161 GFLOP, and 200 recurrence steps h@w_hh^T :
// 80 GFLOP) run on v_wmma_f32_16x16x32_f16 with f32 accumulation.
// Weights pre-converted to f16 once; activations converted f32->f16 through
// LDS; W tiles staged with async global->LDS loads (ASYNCcnt path); C tiles
// staged through LDS for coalesced b128 stores.
// ---------------------------------------------------------------------------

typedef __attribute__((ext_vector_type(16))) _Float16 v16h;
typedef __attribute__((ext_vector_type(8)))  _Float16 v8h;
typedef __attribute__((ext_vector_type(4)))  _Float16 v4h;
typedef __attribute__((ext_vector_type(2)))  _Float16 v2h;
typedef __attribute__((ext_vector_type(8)))  float    v8f;

typedef int v4i __attribute__((ext_vector_type(4)));
typedef __attribute__((address_space(1))) v4i v4i_as1;   // global
typedef __attribute__((address_space(3))) v4i v4i_as3;   // LDS

static constexpr int B_   = 1024;
static constexpr int T_   = 200;
static constexpr int D_   = 512;
static constexpr int H_   = 256;
static constexpr int G_   = 3 * H_;   // 768
static constexpr int KCL  = 12;
static constexpr int KMIT = 30;

#if defined(__HIP_DEVICE_COMPILE__) && \
    __has_builtin(__builtin_amdgcn_global_load_async_to_lds_b128) && \
    __has_builtin(__builtin_amdgcn_s_wait_asynccnt)
#define ASYNC_LDS 1
#else
#define ASYNC_LDS 0
#endif

// ------------------------------- helpers -----------------------------------

__global__ __launch_bounds__(256) void cvt_f32_to_f16(const float* __restrict__ src,
                                                      _Float16* __restrict__ dst, int n) {
  int i = blockIdx.x * 256 + threadIdx.x;
  if (i < n) dst[i] = (_Float16)src[i];
}

__global__ __launch_bounds__(256) void zero_f32(float* __restrict__ p, int n) {
  int i = blockIdx.x * 256 + threadIdx.x;
  if (i < n) p[i] = 0.0f;
}

__device__ __forceinline__ float sigmoidf_(float x) { return 1.0f / (1.0f + __expf(-x)); }

// --------------------------- WMMA GEMM kernel -------------------------------
// C[M,N](f16) = A[M,K](f32, lda=K) * W[N,K]^T(f16) + bias[N](f32)
// Requires M%128==0, N%64==0, K%32==0 (true for all uses here).
// Block: 256 threads = 8 waves laid out 4(M) x 2(N); each wave: 32x32 tile.

static constexpr int BM = 128, BN = 64, BK = 32;
static constexpr int LDA_S = BK + 8;   // A tile row stride (f16), 80B
static constexpr int LDW_S = BK + 16;  // W tile row stride (f16), 96B (16B aligned)
static constexpr int LDC_S = BN + 8;   // C staging row stride (f16), 144B (16B aligned)
static constexpr int SMEM_H = (BM * LDC_S > BM * LDA_S + BN * LDW_S)
                                  ? BM * LDC_S : (BM * LDA_S + BN * LDW_S);

__device__ __forceinline__ v16h frag_from_lds(const _Float16* __restrict__ base,
                                              int ldh, int rbase, int lane) {
  const int half = lane >> 4, lr = lane & 15;
  const _Float16* row = base + (rbase + lr) * ldh;
  v16h f;
#pragma unroll
  for (int j = 0; j < 8; ++j) {
    const int k0 = half * 8 + (j < 4 ? 2 * j : 16 + 2 * (j - 4));
    v2h p = *(const v2h*)(row + k0);
    f[2 * j]     = p[0];
    f[2 * j + 1] = p[1];
  }
  return f;
}

__global__ __launch_bounds__(256)
void gemm_wmma_f16(const float* __restrict__ A, const _Float16* __restrict__ W,
                   const float* __restrict__ bias, _Float16* __restrict__ C,
                   int M, int N, int K) {
  __shared__ _Float16 smem[SMEM_H];
  _Float16* As = smem;                 // BM x LDA_S
  _Float16* Ws = smem + BM * LDA_S;    // BN x LDW_S
  _Float16* Cs = smem;                 // BM x LDC_S (reused after compute)

  const int tid  = threadIdx.x;
  const int wave = tid >> 5;
  const int lane = tid & 31;
  const int wm   = wave & 3;   // 0..3 -> 32-row strip
  const int wn   = wave >> 2;  // 0..1 -> 32-col strip

  const int rowBase = blockIdx.y * BM;
  const int colBase = blockIdx.x * BN;

  v8f acc00 = {}, acc01 = {}, acc10 = {}, acc11 = {};

  for (int k0 = 0; k0 < K; k0 += BK) {
    // prefetch next A K-slice (global_prefetch_b8; fetches whole lines)
    if (k0 + BK < K) {
      __builtin_prefetch(A + (size_t)(rowBase + (tid >> 1)) * K + (k0 + BK) + (tid & 1) * 16, 0, 0);
    }
    __syncthreads();
    // --- stage A: 128x32 f32 -> f16 LDS (1024 float4 chunks / 256 threads)
#pragma unroll
    for (int q = 0; q < 4; ++q) {
      const int c  = tid + 256 * q;     // 0..1023
      const int r  = c >> 3;            // 0..127
      const int cv = c & 7;             // float4 slot
      const float4 v = *(const float4*)(A + (size_t)(rowBase + r) * K + k0 + cv * 4);
      v4h av;
      av[0] = (_Float16)v.x; av[1] = (_Float16)v.y;
      av[2] = (_Float16)v.z; av[3] = (_Float16)v.w;
      *(v4h*)(As + r * LDA_S + cv * 4) = av;
    }
    // --- stage W (already f16, [N,K] row-major): 64x32 tile
#if ASYNC_LDS
    {
      // 256 chunks of 16B: one async b128 per lane (8 wave-instrs per block),
      // tracked with ASYNCcnt (global_load_async_to_lds_b128)
      const int n  = tid >> 2;          // 0..63
      const int kp = (tid & 3) * 8;     // 0,8,16,24 (f16 units)
      v4i_as1* gsrc = (v4i_as1*)(W + (size_t)(colBase + n) * K + k0 + kp);
      v4i_as3* ldst = (v4i_as3*)(Ws + n * LDW_S + kp);
      __builtin_amdgcn_global_load_async_to_lds_b128(gsrc, ldst, 0, 0);
      __builtin_amdgcn_s_wait_asynccnt(0);
    }
#else
#pragma unroll
    for (int q = 0; q < 2; ++q) {
      const int c  = tid + 256 * q;     // 0..511
      const int n  = c >> 3;            // 0..63
      const int kp = (c & 7) * 4;
      v4h wv = *(const v4h*)(W + (size_t)(colBase + n) * K + k0 + kp);
      *(v4h*)(Ws + n * LDW_S + kp) = wv;
    }
#endif
    __syncthreads();

    const v16h a0 = frag_from_lds(As, LDA_S, wm * 32,      lane);
    const v16h a1 = frag_from_lds(As, LDA_S, wm * 32 + 16, lane);
    const v16h b0 = frag_from_lds(Ws, LDW_S, wn * 32,      lane);
    const v16h b1 = frag_from_lds(Ws, LDW_S, wn * 32 + 16, lane);

    acc00 = __builtin_amdgcn_wmma_f32_16x16x32_f16(false, a0, false, b0, (short)0, acc00, false, false);
    acc01 = __builtin_amdgcn_wmma_f32_16x16x32_f16(false, a0, false, b1, (short)0, acc01, false, false);
    acc10 = __builtin_amdgcn_wmma_f32_16x16x32_f16(false, a1, false, b0, (short)0, acc10, false, false);
    acc11 = __builtin_amdgcn_wmma_f32_16x16x32_f16(false, a1, false, b1, (short)0, acc11, false, false);
  }

  // --- epilogue: bias + convert into LDS staging, then coalesced b128 stores
  __syncthreads();   // all LDS tile reads complete before repurposing smem
  {
    const int half = lane >> 4, lr = lane & 15;
    const v8f* accs[4] = { &acc00, &acc01, &acc10, &acc11 };
#pragma unroll
    for (int mi = 0; mi < 2; ++mi) {
#pragma unroll
      for (int ni = 0; ni < 2; ++ni) {
        const v8f& a = *accs[mi * 2 + ni];
        const int Rl = wm * 32 + mi * 16 + half * 8;
        const int Cl = wn * 32 + ni * 16 + lr;
        const float bv = bias[colBase + Cl];
#pragma unroll
        for (int r = 0; r < 8; ++r) {
          Cs[(Rl + r) * LDC_S + Cl] = (_Float16)(a[r] + bv);
        }
      }
    }
  }
  __syncthreads();
  // 128x64 f16 tile = 1024 chunks of 16B; 4 per thread, fully coalesced
#pragma unroll
  for (int q = 0; q < 4; ++q) {
    const int c   = tid + 256 * q;   // 0..1023
    const int row = c >> 3;          // 0..127
    const int col = (c & 7) * 8;     // 0..56
    v8h val = *(const v8h*)(Cs + row * LDC_S + col);
    *(v8h*)(C + (size_t)(rowBase + row) * N + colBase + col) = val;
  }
}

// ------------------------------ GRU gates ----------------------------------

__global__ __launch_bounds__(256)
void gru_gate(const _Float16* __restrict__ xp,   // [B*T, 768] (b_ih fused)
              const _Float16* __restrict__ hp,   // [B, 768]   (b_hh fused)
              float* __restrict__ h,             // [B, 256]
              const int* __restrict__ lengths, int t) {
  const int idx = blockIdx.x * 256 + threadIdx.x;
  const int b = idx >> 8, j = idx & 255;
  if (t >= lengths[b]) return;  // freeze hidden past the sequence length
  const size_t xo = ((size_t)b * T_ + t) * G_;
  const size_t ho = (size_t)b * G_;
  const float xr = (float)xp[xo + j], xz = (float)xp[xo + 256 + j], xn = (float)xp[xo + 512 + j];
  const float hr = (float)hp[ho + j], hz = (float)hp[ho + 256 + j], hn = (float)hp[ho + 512 + j];
  const float r = sigmoidf_(xr + hr);
  const float z = sigmoidf_(xz + hz);
  const float n = tanhf(xn + r * hn);
  const float ho_ = h[(size_t)b * H_ + j];
  h[(size_t)b * H_ + j] = (1.0f - z) * n + z * ho_;
}

// ------------------------------- k-means -----------------------------------

__global__ __launch_bounds__(256)
void km_init(const float* __restrict__ hidden, float* __restrict__ centers) {
  static const int idx[KCL] = {17, 93, 211, 345, 402, 498, 555, 640, 713, 808, 901, 1000};
  centers[blockIdx.x * H_ + threadIdx.x] = hidden[(size_t)idx[blockIdx.x] * H_ + threadIdx.x];
}

__global__ __launch_bounds__(256)
void km_assign(const float* __restrict__ hidden, const float* __restrict__ centers,
               float* __restrict__ sums, float* __restrict__ cnts) {
  __shared__ float c[KCL * H_];
  for (int p = threadIdx.x; p < KCL * H_; p += 256) c[p] = centers[p];
  __syncthreads();
  const int i = blockIdx.x * 256 + threadIdx.x;   // point 0..1023
  const float* hrow = hidden + (size_t)i * H_;
  float best = 3.4e38f; int code = 0;
#pragma unroll 1
  for (int k = 0; k < KCL; ++k) {
    float d = 0.0f;
    for (int hh = 0; hh < H_; ++hh) {
      const float df = hrow[hh] - c[k * H_ + hh];
      d += df * df;
    }
    if (d < best) { best = d; code = k; }
  }
  for (int hh = 0; hh < H_; ++hh) atomicAdd(&sums[code * H_ + hh], hrow[hh]);
  atomicAdd(&cnts[code], 1.0f);
}

__global__ __launch_bounds__(256)
void km_update(const float* __restrict__ sums, const float* __restrict__ cnts,
               float* __restrict__ centers) {
  const int k = blockIdx.x, j = threadIdx.x;
  const float cn = cnts[k];
  if (cn > 0.0f) centers[k * H_ + j] = sums[k * H_ + j] / fmaxf(cn, 1.0f);
}

// ------------------------------- GCN (12x256) ------------------------------

__global__ __launch_bounds__(256)
void gcn_layer(const float* __restrict__ in, const float* __restrict__ w,
               const float* __restrict__ bias, float* __restrict__ out) {
  const int k = blockIdx.x, j = threadIdx.x;
  __shared__ float row[H_];
  row[j] = in[k * H_ + j];
  __syncthreads();
  float acc = bias[j];
  for (int i = 0; i < H_; ++i) acc += row[i] * w[i * H_ + j];
  out[k * H_ + j] = fmaxf(acc, 0.0f);
}

// --------------------------- final fused mixing ----------------------------

__global__ __launch_bounds__(256)
void final_fuse(const float* __restrict__ hidden, const float* __restrict__ centers,
                const float* __restrict__ hprime,
                const float* __restrict__ wt1_w, const float* __restrict__ wt1_b,
                const float* __restrict__ wt2_w, const float* __restrict__ wt2_b,
                float* __restrict__ out) {
  const int b = blockIdx.x, j = threadIdx.x;
  __shared__ float red[256];
  __shared__ float sc[KCL];
  __shared__ float w1s, w2s;
  const float hv = hidden[(size_t)b * H_ + j];

  // e_k = relu(hidden . centers_k), softmax over k
  for (int k = 0; k < KCL; ++k) {
    red[j] = hv * centers[k * H_ + j];
    __syncthreads();
    for (int s = 128; s > 0; s >>= 1) { if (j < s) red[j] += red[j + s]; __syncthreads(); }
    if (j == 0) sc[k] = fmaxf(red[0], 0.0f);
    __syncthreads();
  }
  if (j == 0) {
    float mx = sc[0];
    for (int k = 1; k < KCL; ++k) mx = fmaxf(mx, sc[k]);
    float s = 0.0f;
    for (int k = 0; k < KCL; ++k) { sc[k] = __expf(sc[k] - mx); s += sc[k]; }
    for (int k = 0; k < KCL; ++k) sc[k] /= s;
  }
  __syncthreads();

  float clu = 0.0f;
  for (int k = 0; k < KCL; ++k) clu += sc[k] * hprime[k * H_ + j];

  red[j] = clu * wt1_w[j];
  __syncthreads();
  for (int s = 128; s > 0; s >>= 1) { if (j < s) red[j] += red[j + s]; __syncthreads(); }
  if (j == 0) w1s = sigmoidf_(red[0] + wt1_b[0]);
  __syncthreads();

  red[j] = hv * wt2_w[j];
  __syncthreads();
  for (int s = 128; s > 0; s >>= 1) { if (j < s) red[j] += red[j + s]; __syncthreads(); }
  if (j == 0) w2s = sigmoidf_(red[0] + wt2_b[0]);
  __syncthreads();

  const float w1 = w1s / (w1s + w2s + 1e-8f);
  out[(size_t)b * H_ + j] = w1 * clu + (1.0f - w1) * hv;
}

// ------------------------------- launcher ----------------------------------

extern "C" void kernel_launch(void* const* d_in, const int* in_sizes, int n_in,
                              void* d_out, int out_size, void* d_ws, size_t ws_size,
                              hipStream_t stream) {
  const float* x      = (const float*)d_in[0];
  const int*   lens   = (const int*)  d_in[1];
  const float* w_ih   = (const float*)d_in[2];
  const float* w_hh   = (const float*)d_in[3];
  const float* b_ih   = (const float*)d_in[4];
  const float* b_hh   = (const float*)d_in[5];
  const float* gcn1_w = (const float*)d_in[6];
  const float* gcn1_b = (const float*)d_in[7];
  const float* gcn2_w = (const float*)d_in[8];
  const float* gcn2_b = (const float*)d_in[9];
  const float* wt1_w  = (const float*)d_in[10];
  const float* wt1_b  = (const float*)d_in[11];
  const float* wt2_w  = (const float*)d_in[12];
  const float* wt2_b  = (const float*)d_in[13];
  float* out = (float*)d_out;

  // ---- workspace carve (256B aligned slices) ----
  char* p = (char*)d_ws;
  auto carve = [&p](size_t bytes) -> void* {
    void* r = (void*)p;
    p += (bytes + 255) & ~(size_t)255;
    return r;
  };
  _Float16* wih16   = (_Float16*)carve((size_t)G_ * D_ * sizeof(_Float16));
  _Float16* whh16   = (_Float16*)carve((size_t)G_ * H_ * sizeof(_Float16));
  _Float16* xp      = (_Float16*)carve((size_t)B_ * T_ * G_ * sizeof(_Float16));
  _Float16* hp      = (_Float16*)carve((size_t)B_ * G_ * sizeof(_Float16));
  float*    hbuf    = (float*)carve((size_t)B_ * H_ * sizeof(float));
  float*    centers = (float*)carve((size_t)KCL * H_ * sizeof(float));
  float*    sums    = (float*)carve((size_t)KCL * H_ * sizeof(float));
  float*    cnts    = (float*)carve(64 * sizeof(float));
  float*    gtmp    = (float*)carve((size_t)KCL * H_ * sizeof(float));
  float*    hprime  = (float*)carve((size_t)KCL * H_ * sizeof(float));

  // ---- weight conversion + hidden init ----
  cvt_f32_to_f16<<<(G_ * D_ + 255) / 256, 256, 0, stream>>>(w_ih, wih16, G_ * D_);
  cvt_f32_to_f16<<<(G_ * H_ + 255) / 256, 256, 0, stream>>>(w_hh, whh16, G_ * H_);
  zero_f32<<<(B_ * H_ + 255) / 256, 256, 0, stream>>>(hbuf, B_ * H_);

  // ---- x_proj = x @ w_ih^T + b_ih : [204800, 768], K=512 ----
  {
    dim3 grid(G_ / BN, (B_ * T_) / BM);
    gemm_wmma_f16<<<grid, 256, 0, stream>>>(x, wih16, b_ih, xp, B_ * T_, G_, D_);
  }

  // ---- GRU recurrence: 200 sequential steps ----
  {
    dim3 grid(G_ / BN, B_ / BM);
    for (int t = 0; t < T_; ++t) {
      gemm_wmma_f16<<<grid, 256, 0, stream>>>(hbuf, whh16, b_hh, hp, B_, G_, H_);
      gru_gate<<<(B_ * H_) / 256, 256, 0, stream>>>(xp, hp, hbuf, lens, t);
    }
  }

  // ---- k-means (30 iterations) ----
  km_init<<<KCL, 256, 0, stream>>>(hbuf, centers);
  for (int it = 0; it < KMIT; ++it) {
    zero_f32<<<(KCL * H_ + 255) / 256, 256, 0, stream>>>(sums, KCL * H_);
    zero_f32<<<1, 256, 0, stream>>>(cnts, 64);
    km_assign<<<B_ / 256, 256, 0, stream>>>(hbuf, centers, sums, cnts);
    km_update<<<KCL, 256, 0, stream>>>(sums, cnts, centers);
  }

  // ---- GCN (identity adjacency => per-center MLP) ----
  gcn_layer<<<KCL, 256, 0, stream>>>(centers, gcn1_w, gcn1_b, gtmp);
  gcn_layer<<<KCL, 256, 0, stream>>>(gtmp, gcn2_w, gcn2_b, hprime);

  // ---- fused scores/softmax/mixing ----
  final_fuse<<<B_, 256, 0, stream>>>(hbuf, centers, hprime,
                                     wt1_w, wt1_b, wt2_w, wt2_b, out);
}
